// MultiHeadAttention_30666066494093
// MI455X (gfx1250) — matrix operand order repaired and checked by reading, not compiled, not run on hardware
//
#include <hip/hip_runtime.h>

// ---------------------------------------------------------------------------
// MHA forward for MI455X (gfx1250, wave32, WMMA bf16 f32-accumulate).
// Pipeline: convert -> QKV GEMMs (WMMA, 32x32/wave) ->
//           flash attention (WMMA + TDM double-buffered LDS tiles) -> out GEMM.
// Confirmed lowerings: v_wmma_f32_16x16x32_bf16, tensor_load_to_lds,
// global_prefetch_b8, s_wait_tensorcnt/dscnt, s_clause'd global_load_b128.
// ---------------------------------------------------------------------------

typedef __attribute__((ext_vector_type(16))) __bf16 v16bf;
typedef __attribute__((ext_vector_type(8)))  float  v8f;
typedef __attribute__((ext_vector_type(4)))  unsigned int u32x4;
typedef __attribute__((ext_vector_type(8)))  int i32x8;
typedef __attribute__((ext_vector_type(4)))  int i32x4;

#if defined(__has_builtin)
#  if __has_builtin(__builtin_amdgcn_tensor_load_to_lds)
#    define ATHENA_HAVE_TDM 1
#  endif
#endif

union Frag {
    unsigned int u[8];
    v16bf v;
};

static __device__ __forceinline__ unsigned short f2bf(float f) {
    unsigned int u = __builtin_bit_cast(unsigned int, f);
    unsigned int r = (u + 0x7FFFu + ((u >> 16) & 1u)) >> 16;  // RNE
    return (unsigned short)r;
}

static __device__ __forceinline__ v8f wmma_bf16(const Frag& a, const Frag& b, v8f c) {
    return __builtin_amdgcn_wmma_f32_16x16x32_bf16(
        false, a.v, false, b.v, (short)0, c, false, false);
}

#ifdef ATHENA_HAVE_TDM
// -------------------------------------------------------------------------
// Issue a 2-D TDM tile load global->LDS (TENSOR_LOAD_TO_LDS).
// D# group0: count=1, lds_addr, global_addr, type=2.
// D# group1: data_size=2B, pad_enable + pad_interval/amount (hardware LDS
// row padding == our bank-conflict-free strides), tensor/tile dims, stride0.
// Tracked on TENSORcnt; completion via s_wait_tensorcnt.
// -------------------------------------------------------------------------
static __device__ __forceinline__ void tdm_load_2d(
        const void* gptr, unsigned lds_off,
        unsigned tensor_d0, unsigned tensor_d1,
        unsigned tile_d0, unsigned tile_d1,
        unsigned long long stride0_elts,
        unsigned pad_interval, unsigned pad_amount) {
    unsigned long long ga = (unsigned long long)gptr;
    u32x4 g0 = {};
    g0[0] = 1u;                                                 // count = 1 descriptor
    g0[1] = lds_off;                                            // LDS byte address
    g0[2] = (unsigned)ga;                                       // global addr lo
    g0[3] = (unsigned)((ga >> 32) & 0x01FFFFFFu) | (2u << 30);  // addr hi | type=2
    i32x8 g1 = {};
    g1[0] = (int)((1u << 16)                 // data_size = 1 -> 2 bytes
                  | (1u << 20)               // pad_enable
                  | (pad_interval << 22)     // dwords before pad (log enc)
                  | (pad_amount << 25));     // pad dwords - 1
    g1[1] = (int)((tensor_d0 & 0xFFFFu) << 16);
    g1[2] = (int)((tensor_d0 >> 16) | ((tensor_d1 & 0xFFFFu) << 16));
    g1[3] = (int)((tensor_d1 >> 16) | (tile_d0 << 16));
    g1[4] = (int)(tile_d1 & 0xFFFFu);        // tile_dim1 (tile_dim2 = 0)
    g1[5] = (int)(unsigned)(stride0_elts & 0xFFFFFFFFull);
    g1[6] = (int)(unsigned)((stride0_elts >> 32) & 0xFFFFull);
    g1[7] = 0;
    i32x4 z4 = {};
#if __clang_major__ >= 23
    i32x8 z8 = {};
    __builtin_amdgcn_tensor_load_to_lds(g0, g1, z4, z4, z8, 0);
#else
    __builtin_amdgcn_tensor_load_to_lds(g0, g1, z4, z4, 0);
#endif
}
#endif  // ATHENA_HAVE_TDM

// ---------------------------------------------------------------------------
// Elementwise fp32 -> bf16, and transposing weight convert.
// ---------------------------------------------------------------------------
__global__ void cvt_f32_bf16(const float* __restrict__ src,
                             unsigned short* __restrict__ dst, int n) {
    int i = blockIdx.x * blockDim.x + threadIdx.x;
    if (i < n) dst[i] = f2bf(src[i]);
}

__global__ void transpose_w_bf16(const float* __restrict__ w,
                                 unsigned short* __restrict__ wt) {
    int i = blockIdx.x * blockDim.x + threadIdx.x;  // 512*512
    int n = i >> 9, k = i & 511;
    wt[i] = f2bf(w[k * 512 + n]);                   // wt[n][k] = w[k][n]
}

// ---------------------------------------------------------------------------
// bf16 WMMA GEMM:  Y[4096,512] = A[4096,512] * Wt^T + bias
// One wave computes a 32x32 tile (4 accumulators, 2 A-frags x 2 B-frags
// reused -> 2x arithmetic intensity). Block = 4 waves in 2x2 -> 64x64 tile.
//   mode 0: f32 row-major   mode 1: bf16 (b,h,s,d)   mode 2: bf16 (b,h,d,s)
// ---------------------------------------------------------------------------
static __device__ __forceinline__ void store_tile(
        v8f acc, int mt, int nt, int ln, int half, float bv, int mode,
        float* __restrict__ outF, unsigned short* __restrict__ outB) {
    const int n = nt + ln;
#pragma unroll
    for (int r = 0; r < 8; ++r) {
        int m = mt + r + 8 * half;              // C/D layout row
        float val = acc[r] + bv;
        if (mode == 0) {
            outF[(size_t)m * 512 + n] = val;
        } else {
            int bb = m >> 11, s = m & 2047;     // m = b*2048 + s
            int hh = n >> 6,  d = n & 63;       // n = h*64 + d
            if (mode == 1)
                outB[(((size_t)(bb * 8 + hh)) * 2048 + s) * 64 + d] = f2bf(val);
            else
                outB[(((size_t)(bb * 8 + hh)) * 64 + d) * 2048 + s] = f2bf(val);
        }
    }
}

__global__ __launch_bounds__(128) void gemm_bf16(
        const unsigned short* __restrict__ A,
        const unsigned short* __restrict__ Wt,
        const float* __restrict__ bias,
        float* __restrict__ outF,
        unsigned short* __restrict__ outB,
        int mode) {
    const int lane = threadIdx.x & 31, wave = threadIdx.x >> 5;
    const int half = lane >> 4, ln = lane & 15;
    const int m0 = blockIdx.x * 64 + (wave >> 1) * 32;
    const int n0 = blockIdx.y * 64 + (wave & 1) * 32;

    const unsigned short* a0p = A  + (size_t)(m0 + ln) * 512;
    const unsigned short* a1p = A  + (size_t)(m0 + 16 + ln) * 512;
    const unsigned short* b0p = Wt + (size_t)(n0 + ln) * 512;
    const unsigned short* b1p = Wt + (size_t)(n0 + 16 + ln) * 512;

    v8f acc00 = {}, acc01 = {}, acc10 = {}, acc11 = {};
    for (int kk = 0; kk < 512; kk += 32) {
        // locality 3 -> WGP-scope prefetch (populate all cache levels),
        // not SYS scope (GL2-only) as locality 0 produced.
        __builtin_prefetch(a0p + kk + 256, 0, 3);   // -> global_prefetch_b8
        __builtin_prefetch(b0p + kk + 256, 0, 3);
        Frag a0, a1, b0, b1;
#pragma unroll
        for (int r = 0; r < 8; ++r) {
            // 16-bit A-layout pair base, same k order for A and B (ISA 7.12.2)
            int ka = kk + 2 * r + (r >= 4 ? 8 : 0) + 8 * half;
            int kb = kk + 2 * r + (r >= 4 ? 8 : 0) + 8 * half;
            a0.u[r] = *(const unsigned int*)(a0p + ka);
            a1.u[r] = *(const unsigned int*)(a1p + ka);
            b0.u[r] = *(const unsigned int*)(b0p + kb);
            b1.u[r] = *(const unsigned int*)(b1p + kb);
        }
        acc00 = wmma_bf16(a0, b0, acc00);
        acc01 = wmma_bf16(a0, b1, acc01);
        acc10 = wmma_bf16(a1, b0, acc10);
        acc11 = wmma_bf16(a1, b1, acc11);
    }

    const float bv0 = bias[n0 + ln];
    const float bv1 = bias[n0 + 16 + ln];
    store_tile(acc00, m0,      n0,      ln, half, bv0, mode, outF, outB);
    store_tile(acc01, m0,      n0 + 16, ln, half, bv1, mode, outF, outB);
    store_tile(acc10, m0 + 16, n0,      ln, half, bv0, mode, outF, outB);
    store_tile(acc11, m0 + 16, n0 + 16, ln, half, bv1, mode, outF, outB);
}

// ---------------------------------------------------------------------------
// Flash attention: grid (S/64, H, B), block = 128 = 4 waves.
// Wave w handles queries [q0, q0+16). Keys stream in tiles of 32 through LDS,
// double-buffered: wave0/wave1 issue TDM tile DMAs (K and V^T resp.) for
// tile kt+1 while all waves compute tile kt. TENSORcnt + barrier publishes.
// ---------------------------------------------------------------------------
__global__ __launch_bounds__(128) void attn_fwd(
        const unsigned short* __restrict__ Qh,
        const unsigned short* __restrict__ Kh,
        const unsigned short* __restrict__ Vt,
        const float* __restrict__ mask,
        unsigned short* __restrict__ merged) {
    __shared__ unsigned int ldsK[2][32 * 36];   // 32 keys x 64 bf16, stride 36 dw
    __shared__ unsigned int ldsV[2][64 * 20];   // 64 dims x 32 bf16, stride 20 dw
    __shared__ unsigned int ldsP[4][16 * 17];   // per-wave P 16x32 bf16, stride 17 dw

    const int tid = threadIdx.x, lane = tid & 31, wave = tid >> 5;
    const int half = lane >> 4, ln = lane & 15;
    const int b = blockIdx.z, h = blockIdx.y;
    const int q0 = blockIdx.x * 64 + wave * 16;
    const size_t headR = ((size_t)(b * 8 + h)) * 2048;  // rows-of-64 base (Qh/Kh)
    const size_t headV = ((size_t)(b * 8 + h)) * 64;    // rows-of-2048 base (Vt)
    const unsigned short* kbase = Kh + headR * 64;
    const unsigned short* vbase = Vt + headV * 2048;

    // Q fragments (16x64 = two 16x32 A-frags), resident for the whole loop.
    Frag qf[2];
#pragma unroll
    for (int c = 0; c < 2; ++c)
#pragma unroll
        for (int r = 0; r < 8; ++r) {
            int ka = c * 32 + 2 * r + (r >= 4 ? 8 : 0) + 8 * half;
            qf[c].u[r] = *(const unsigned int*)(Qh + (headR + q0 + ln) * 64 + ka);
        }

    v8f o0 = {}, o1 = {}, o2 = {}, o3 = {};
    float rmax[8], rsum[8];
#pragma unroll
    for (int r = 0; r < 8; ++r) { rmax[r] = -3.0e38f; rsum[r] = 0.f; }

#ifdef ATHENA_HAVE_TDM
    const unsigned offK0 = (unsigned)(unsigned long long)(const void*)&ldsK[0][0];
    const unsigned offK1 = (unsigned)(unsigned long long)(const void*)&ldsK[1][0];
    const unsigned offV0 = (unsigned)(unsigned long long)(const void*)&ldsV[0][0];
    const unsigned offV1 = (unsigned)(unsigned long long)(const void*)&ldsV[1][0];
    // Prime buffer 0: K tile 32x64 (pad 32dw->+4dw), V^T tile 64x32 (16dw->+4dw)
    if (wave == 0)
        tdm_load_2d(kbase, offK0, 64, 2048, 64, 32, 64, /*int*/4, /*amt*/3);
    if (wave == 1)
        tdm_load_2d(vbase, offV0, 2048, 64, 32, 64, 2048, /*int*/3, /*amt*/3);
#endif

    for (int kt = 0; kt < 64; ++kt) {
        const int cur = kt & 1;
#ifdef ATHENA_HAVE_TDM
        if (kt + 1 < 64) {
            if (wave == 0)
                tdm_load_2d(kbase + (size_t)(kt + 1) * 32 * 64,
                            cur ? offK0 : offK1, 64, 2048, 64, 32, 64, 4, 3);
            if (wave == 1)
                tdm_load_2d(vbase + (size_t)(kt + 1) * 32,
                            cur ? offV0 : offV1, 2048, 64, 32, 64, 2048, 3, 3);
            if (wave < 2) __builtin_amdgcn_s_wait_tensorcnt(1);  // cur tile done
        } else {
            if (wave < 2) __builtin_amdgcn_s_wait_tensorcnt(0);
        }
        __syncthreads();   // publish DMA completion to all waves
#else
        for (int i = tid; i < 32 * 32; i += 128) {
            int rr = i >> 5, cc = i & 31;
            ldsK[cur][rr * 36 + cc] =
                *(const unsigned int*)(kbase + ((size_t)kt * 32 + rr) * 64 + cc * 2);
        }
        for (int i = tid; i < 64 * 16; i += 128) {
            int rr = i >> 4, cc = i & 15;
            ldsV[cur][rr * 20 + cc] =
                *(const unsigned int*)(vbase + (size_t)rr * 2048 + kt * 32 + cc * 2);
        }
        __syncthreads();
#endif

        // Scores: S[16q x 32k] as two 16x16 accumulators over K-dim DH=64.
        v8f s0 = {}, s1 = {};
#pragma unroll
        for (int c = 0; c < 2; ++c) {
            Frag b0, b1;
#pragma unroll
            for (int r = 0; r < 8; ++r) {
                int dw = c * 16 + r + (r >= 4 ? 4 : 0) + 4 * half;
                b0.u[r] = ldsK[cur][ln * 36 + dw];
                b1.u[r] = ldsK[cur][(16 + ln) * 36 + dw];
            }
            s0 = wmma_bf16(qf[c], b0, s0);
            s1 = wmma_bf16(qf[c], b1, s1);
        }

        // Scale + mask + online softmax (C-layout rows: m = r + 8*half).
        unsigned short* Pw = (unsigned short*)(&ldsP[wave][0]);
        float ps[8];
#pragma unroll
        for (int r = 0; r < 8; ++r) {
            int m = r + 8 * half;
            size_t mrow = ((size_t)b * 2048 + (q0 + m)) * 2048 + (size_t)kt * 32;
            float f0 = s0[r] * 0.125f - 1.0e9f * mask[mrow + ln];
            float f1 = s1[r] * 0.125f - 1.0e9f * mask[mrow + 16 + ln];
            float mx = fmaxf(f0, f1);
#pragma unroll
            for (int off = 1; off < 16; off <<= 1)
                mx = fmaxf(mx, __shfl_xor(mx, off, 16));
            float nm = fmaxf(rmax[r], mx);
            float sc = __expf(rmax[r] - nm);
            rmax[r] = nm;
            float e0 = __expf(f0 - nm), e1 = __expf(f1 - nm);
            ps[r] = e0 + e1;
            rsum[r] *= sc;
            o0[r] *= sc; o1[r] *= sc; o2[r] *= sc; o3[r] *= sc;
            Pw[m * 34 + ln]      = f2bf(e0);   // C-layout -> A-layout bounce
            Pw[m * 34 + 16 + ln] = f2bf(e1);
        }
#pragma unroll
        for (int r = 0; r < 8; ++r) {
            float p = ps[r];
#pragma unroll
            for (int off = 1; off < 16; off <<= 1) p += __shfl_xor(p, off, 16);
            rsum[r] += p;
        }
        asm volatile("s_wait_dscnt 0" ::: "memory");  // own-wave LDS RAW

        Frag pa;
#pragma unroll
        for (int r = 0; r < 8; ++r)
            pa.u[r] = ldsP[wave][ln * 17 + r + (r >= 4 ? 4 : 0) + 4 * half];

        Frag v0, v1, v2, v3;
#pragma unroll
        for (int r = 0; r < 8; ++r) {
            int dw = r + (r >= 4 ? 4 : 0) + 4 * half;
            v0.u[r] = ldsV[cur][(ln)      * 20 + dw];
            v1.u[r] = ldsV[cur][(16 + ln) * 20 + dw];
            v2.u[r] = ldsV[cur][(32 + ln) * 20 + dw];
            v3.u[r] = ldsV[cur][(48 + ln) * 20 + dw];
        }
        o0 = wmma_bf16(pa, v0, o0);
        o1 = wmma_bf16(pa, v1, o1);
        o2 = wmma_bf16(pa, v2, o2);
        o3 = wmma_bf16(pa, v3, o3);
        __syncthreads();   // readers done before next DMA overwrites buffer
    }

    // Normalize and write merged (B,S,512) bf16.
#pragma unroll
    for (int r = 0; r < 8; ++r) {
        int m = r + 8 * half;
        float inv = 1.0f / rsum[r];
        size_t base = ((size_t)b * 2048 + q0 + m) * 512 + h * 64;
        merged[base + ln]      = f2bf(o0[r] * inv);
        merged[base + 16 + ln] = f2bf(o1[r] * inv);
        merged[base + 32 + ln] = f2bf(o2[r] * inv);
        merged[base + 48 + ln] = f2bf(o3[r] * inv);
    }
}

// ---------------------------------------------------------------------------
// Launch
// ---------------------------------------------------------------------------
extern "C" void kernel_launch(void* const* d_in, const int* in_sizes, int n_in,
                              void* d_out, int out_size, void* d_ws, size_t ws_size,
                              hipStream_t stream) {
    const float* q    = (const float*)d_in[0];
    const float* k    = (const float*)d_in[1];
    const float* v    = (const float*)d_in[2];
    const float* mask = (const float*)d_in[3];
    const float* wq   = (const float*)d_in[4];
    const float* bq   = (const float*)d_in[5];
    const float* wk   = (const float*)d_in[6];
    const float* bk   = (const float*)d_in[7];
    const float* wv   = (const float*)d_in[8];
    const float* bv   = (const float*)d_in[9];
    const float* wo   = (const float*)d_in[10];
    const float* bo   = (const float*)d_in[11];

    char* ws = (char*)d_ws;
    const size_t XB = (size_t)4096 * 512 * 2;  // 4 MB bf16 activations
    const size_t WB = (size_t)512 * 512 * 2;   // 512 KB bf16 weights
    unsigned short* xq  = (unsigned short*)(ws);
    unsigned short* xk  = (unsigned short*)(ws + XB);
    unsigned short* xv  = (unsigned short*)(ws + 2 * XB);
    unsigned short* wtq = (unsigned short*)(ws + 3 * XB);
    unsigned short* wtk = (unsigned short*)(ws + 3 * XB + WB);
    unsigned short* wtv = (unsigned short*)(ws + 3 * XB + 2 * WB);
    unsigned short* wto = (unsigned short*)(ws + 3 * XB + 3 * WB);
    unsigned short* qh  = (unsigned short*)(ws + 3 * XB + 4 * WB);
    unsigned short* kh  = (unsigned short*)(ws + 4 * XB + 4 * WB);
    unsigned short* vt  = (unsigned short*)(ws + 5 * XB + 4 * WB);
    unsigned short* mg  = (unsigned short*)(ws + 6 * XB + 4 * WB);
    // total workspace: 7*4MB + 4*0.5MB = 30 MB

    const int nx = 4096 * 512;
    cvt_f32_bf16<<<nx / 256, 256, 0, stream>>>(q, xq, nx);
    cvt_f32_bf16<<<nx / 256, 256, 0, stream>>>(k, xk, nx);
    cvt_f32_bf16<<<nx / 256, 256, 0, stream>>>(v, xv, nx);

    const int nw = 512 * 512;
    transpose_w_bf16<<<nw / 256, 256, 0, stream>>>(wq, wtq);
    transpose_w_bf16<<<nw / 256, 256, 0, stream>>>(wk, wtk);
    transpose_w_bf16<<<nw / 256, 256, 0, stream>>>(wv, wtv);
    transpose_w_bf16<<<nw / 256, 256, 0, stream>>>(wo, wto);

    dim3 gg(4096 / 64, 512 / 64);
    gemm_bf16<<<gg, 128, 0, stream>>>(xq, wtq, bq, nullptr, qh, 1);
    gemm_bf16<<<gg, 128, 0, stream>>>(xk, wtk, bk, nullptr, kh, 1);
    gemm_bf16<<<gg, 128, 0, stream>>>(xv, wtv, bv, nullptr, vt, 2);

    attn_fwd<<<dim3(32, 8, 2), 128, 0, stream>>>(qh, kh, vt, mask, mg);

    gemm_bf16<<<gg, 128, 0, stream>>>(mg, wto, bo, (float*)d_out, nullptr, 0);
}
